// Codebook_54949811585631
// MI455X (gfx1250) — matrix operand order repaired and checked by reading, not compiled, not run on hardware
//
#include <hip/hip_runtime.h>
#include <hip/hip_bf16.h>
#include <stdint.h>

// ---------------- problem constants ----------------
#define NB   32
#define HH   64
#define WW   64
#define DD   64
#define KK   1024
#define NTOK (NB*HH*WW)          // 131072
#define BETA 0.25f
#define DECAY 0.99f
#define EPSV 1e-05f

// ---------------- WMMA types ----------------
typedef __bf16 bf16_t;
typedef __attribute__((ext_vector_type(16))) __bf16 v16bf;
typedef __attribute__((ext_vector_type(8)))  float  v8f;

union Frag32B { uint4 q[2]; v16bf v; };

__device__ __forceinline__ unsigned short f2bf(float f) {
    unsigned int u = __float_as_uint(f);
    u += 0x7FFFu + ((u >> 16) & 1u);   // round-to-nearest-even
    return (unsigned short)(u >> 16);
}

// ---------------- K_init: zero scratch accumulators ----------------
__global__ void k_init(float* __restrict__ cluster_size,
                       float* __restrict__ embed_sum,
                       float* __restrict__ loss_acc) {
    int id = blockIdx.x * 256 + threadIdx.x;
    if (id < KK)       cluster_size[id] = 0.0f;
    if (id < KK * DD)  embed_sum[id]    = 0.0f;
    if (id == 0)       loss_acc[0]      = 0.0f;
}

// ---------------- K_prep_z: [B,C,H,W] f32 -> [N,D] bf16 tokens ----------------
// one block per (b,h); LDS 64x64 transpose tile; both global phases coalesced.
__global__ void k_prep_z(const float* __restrict__ z,
                         unsigned short* __restrict__ flat) {
    __shared__ float tile[DD][WW + 1];
    const int b = blockIdx.x >> 6;
    const int h = blockIdx.x & 63;
    const int t = threadIdx.x;

    // read: z[b][d][h][w], w fastest across lanes
    {
        const int w  = t & 63;
        const int dq = t >> 6;                 // 0..3
        const size_t base = (size_t)b * (DD * HH * WW) + (size_t)h * WW;
#pragma unroll
        for (int j = 0; j < 16; ++j) {
            const int d = dq + 4 * j;
            tile[d][w] = z[base + (size_t)d * (HH * WW) + w];
        }
    }
    __syncthreads();
    // write: flat[(nbase+w)*64 + d], d fastest across lanes (128B rows)
    {
        const int d  = t & 63;
        const int wq = t >> 6;
        const size_t nbase = ((size_t)b * HH + h) * WW;
#pragma unroll
        for (int j = 0; j < 16; ++j) {
            const int w = wq + 4 * j;
            flat[(nbase + w) * DD + d] = f2bf(tile[d][w]);
        }
    }
}

// ---------------- K_prep_e: embedding f32 -> bf16, plus w2[k] ----------------
// one wave per code row
__global__ void k_prep_e(const float* __restrict__ emb,
                         unsigned short* __restrict__ embb,
                         float* __restrict__ w2) {
    const int k = blockIdx.x;
    const int l = threadIdx.x;                 // 0..31
    const float v0 = emb[k * DD + l];
    const float v1 = emb[k * DD + l + 32];
    embb[k * DD + l]      = f2bf(v0);
    embb[k * DD + l + 32] = f2bf(v1);
    float s = v0 * v0 + v1 * v1;
#pragma unroll
    for (int m = 16; m >= 1; m >>= 1) s += __shfl_xor(s, m, 32);
    if (l == 0) w2[k] = s;
}

// ---------------- K_argmin: WMMA bf16 GEMM tiles + running argmin ----------------
// block = 256 threads = 8 waves; each wave owns a 16-token tile.
// 16 stages; each stage stages 64 codes (8 KB) in LDS for all 8 waves and
// runs 4 sub-tiles x 2 WMMAs per barrier pair.
__global__ void k_argmin(const unsigned short* __restrict__ flat,
                         const unsigned short* __restrict__ embb,
                         const float* __restrict__ w2,
                         float* __restrict__ cluster_size,
                         int*   __restrict__ idx_i32,
                         float* __restrict__ idx_f) {
    __shared__ uint4 sB[64 * 8];               // 64 codes x 64 bf16 = 8 KB

    const int t    = threadIdx.x;
    const int lane = t & 31;
    const int wv   = t >> 5;
    const int half = lane >> 4;                // lane group 0/1
    const int col  = lane & 15;

    const int tokBase = (blockIdx.x * 8 + wv) * 16;

    // ---- A fragments (held for whole loop). 16-bit A layout (ISA 7.12.2):
    // lanes 0-15 hold K{0..7,16..23}, lanes 16-31 hold K{8..15,24..31}.
    const uint4* ap = (const uint4*)(flat + (size_t)(tokBase + col) * DD);
    Frag32B a0, a1;
    a0.q[0] = ap[half];     a0.q[1] = ap[2 + half];   // K 0..31
    a1.q[0] = ap[4 + half]; a1.q[1] = ap[6 + half];   // K 32..63

    float minv[8];
    int   mini[8];
#pragma unroll
    for (int i = 0; i < 8; ++i) { minv[i] = __builtin_inff(); mini[i] = 0; }

    // stage = 64 code rows = 1024 uint2 (contiguous in emb_bf16)
    const uint2* emb2 = (const uint2*)embb;
    uint2 pre[4];
#pragma unroll
    for (int p = 0; p < 4; ++p) pre[p] = emb2[t + 256 * p];

    for (int s = 0; s < 16; ++s) {
        __syncthreads();                       // prior stage fully consumed
#pragma unroll
        for (int p = 0; p < 4; ++p) ((uint2*)sB)[t + 256 * p] = pre[p];
        if (s + 1 < 16) {
#pragma unroll
            for (int p = 0; p < 4; ++p)
                pre[p] = emb2[(size_t)(s + 1) * 1024 + t + 256 * p];
        }
        __syncthreads();                       // stage visible to all waves

#pragma unroll
        for (int sub = 0; sub < 4; ++sub) {
            // ---- B fragments (same K layout as A):
            // lanes 0-15 hold K{0..7,16..23}, lanes 16-31 K{8..15,24..31}; column = lane&15.
            const uint4* bp = &sB[(sub * 16 + col) * 8];
            Frag32B b0, b1;
            b0.q[0] = bp[half];     b0.q[1] = bp[2 + half];     // d 0..31
            b1.q[0] = bp[4 + half]; b1.q[1] = bp[6 + half];     // d 32..63

            v8f acc = {};
            acc = __builtin_amdgcn_wmma_f32_16x16x32_bf16(
                      false, a0.v, false, b0.v, (short)0, acc, false, false);
            acc = __builtin_amdgcn_wmma_f32_16x16x32_bf16(
                      false, a1.v, false, b1.v, (short)0, acc, false, false);

            const int   code = (s * 4 + sub) * 16 + col;
            const float wl   = w2[code];
#pragma unroll
            for (int i = 0; i < 8; ++i) {
                const float sc = __builtin_fmaf(-2.0f, acc[i], wl); // dist - x2
                if (sc < minv[i]) { minv[i] = sc; mini[i] = code; }
            }
        }
    }

    // ---- cross-lane argmin over the 16 columns of each half-group.
#pragma unroll
    for (int i = 0; i < 8; ++i) {
        float v  = minv[i];
        int   ix = mini[i];
#pragma unroll
        for (int m = 1; m < 16; m <<= 1) {
            const float ov = __shfl_xor(v, m, 32);
            const int   oi = __shfl_xor(ix, m, 32);
            if (ov < v || (ov == v && oi < ix)) { v = ov; ix = oi; }
        }
        minv[i] = v; mini[i] = ix;
    }

    if (col < 8) {
        int ix = mini[0];
#pragma unroll
        for (int i = 1; i < 8; ++i) if (col == i) ix = mini[i];
        const int row = col + half * 8;        // C layout: slot i -> M=i / M=8+i
        const int tok = tokBase + row;
        idx_i32[tok] = ix;
        idx_f[tok]   = (float)ix;
        atomicAdd(&cluster_size[ix], 1.0f);
    }
}

// ---------------- K_gather: z_q (NCHW), loss partial, embed_sum ----------------
__global__ void k_gather(const float* __restrict__ z,
                         const float* __restrict__ emb,
                         const int*   __restrict__ idx,
                         float* __restrict__ zq,
                         float* __restrict__ embed_sum,
                         float* __restrict__ loss_acc) {
    __shared__ float red[256];
    const int b = blockIdx.x >> 6;
    const int h = blockIdx.x & 63;
    const int t = threadIdx.x;
    const int w  = t & 63;
    const int dq = t >> 6;
    const int n  = (b * HH + h) * WW + w;
    const int k  = idx[n];

    float accl = 0.0f;
#pragma unroll
    for (int j = 0; j < 16; ++j) {
        const int d = dq + 4 * j;
        const float e = emb[k * DD + d];
        const size_t o = (size_t)b * (DD * HH * WW) + (size_t)d * (HH * WW)
                       + (size_t)h * WW + w;
        const float zv = z[o];
        zq[o] = e;                                  // straight-through value == z_q
        atomicAdd(&embed_sum[k * DD + d], zv);
        const float df = zv - e;
        accl += df * df;
    }
    red[t] = accl;
    __syncthreads();
#pragma unroll
    for (int s = 128; s >= 1; s >>= 1) {
        if (t < s) red[t] += red[t + s];
        __syncthreads();
    }
    if (t == 0) atomicAdd(loss_acc, red[0]);
}

// ---------------- K_final: EMA updates, normalization, loss ----------------
__global__ void k_final(const float* __restrict__ ema_cs_old,
                        const float* __restrict__ ema_w_old,
                        const float* __restrict__ cluster_size,
                        const float* __restrict__ embed_sum,
                        const float* __restrict__ loss_acc,
                        float* __restrict__ out_loss,
                        float* __restrict__ out_new_emb,
                        float* __restrict__ out_ema_cs,
                        float* __restrict__ out_ema_w) {
    __shared__ float red[KK];
    const int k = threadIdx.x;
    const float ecs = DECAY * ema_cs_old[k] + (1.0f - DECAY) * cluster_size[k];
    out_ema_cs[k] = ecs;
    red[k] = ecs;
    __syncthreads();
#pragma unroll
    for (int s = 512; s >= 1; s >>= 1) {
        if (k < s) red[k] += red[k + s];
        __syncthreads();
    }
    const float nsum = red[0];
    const float cs   = (ecs + EPSV) / (nsum + (float)KK * EPSV) * nsum;
    const float inv  = 1.0f / cs;
#pragma unroll 4
    for (int d = 0; d < DD; ++d) {
        const float wsv = DECAY * ema_w_old[k * DD + d]
                        + (1.0f - DECAY) * embed_sum[k * DD + d];
        out_ema_w[k * DD + d]   = wsv;
        out_new_emb[k * DD + d] = wsv * inv;
    }
    if (k == 0)
        out_loss[0] = BETA * loss_acc[0] / (float)(NTOK * DD);
}

// ---------------- host launcher ----------------
extern "C" void kernel_launch(void* const* d_in, const int* in_sizes, int n_in,
                              void* d_out, int out_size, void* d_ws, size_t ws_size,
                              hipStream_t stream) {
    const float* z      = (const float*)d_in[0];   // [32,64,64,64]
    const float* emb    = (const float*)d_in[1];   // [1024,64]
    const float* ema_cs = (const float*)d_in[2];   // [1024]
    const float* ema_w  = (const float*)d_in[3];   // [1024,64]

    float* out = (float*)d_out;
    float* out_zq      = out;                              // 8388608
    float* out_idx     = out + 8388608;                    // 131072
    float* out_loss    = out + 8519680;                    // 1
    float* out_new_emb = out + 8519681;                    // 65536
    float* out_ema_cs  = out + 8585217;                    // 1024
    float* out_ema_w   = out + 8586241;                    // 65536

    char* ws = (char*)d_ws;
    unsigned short* flat_bf16 = (unsigned short*)(ws + 0);           // 16 MB
    unsigned short* emb_bf16  = (unsigned short*)(ws + 16777216);    // 128 KB
    float* w2           = (float*)(ws + 16908288);                   // 4 KB
    float* cluster_size = (float*)(ws + 16912384);                   // 4 KB
    float* embed_sum    = (float*)(ws + 16916480);                   // 256 KB
    int*   idx_i32      = (int*)  (ws + 17178624);                   // 512 KB
    float* loss_acc     = (float*)(ws + 17702912);                   // 4 B

    k_init  <<<256, 256, 0, stream>>>(cluster_size, embed_sum, loss_acc);
    k_prep_z<<<NB * HH, 256, 0, stream>>>(z, flat_bf16);
    k_prep_e<<<KK, 32, 0, stream>>>(emb, emb_bf16, w2);
    k_argmin<<<NTOK / 128, 256, 0, stream>>>(flat_bf16, emb_bf16, w2,
                                             cluster_size, idx_i32, out_idx);
    k_gather<<<NB * HH, 256, 0, stream>>>(z, emb, idx_i32, out_zq,
                                          embed_sum, loss_acc);
    k_final <<<1, KK, 0, stream>>>(ema_cs, ema_w, cluster_size, embed_sum,
                                   loss_acc, out_loss, out_new_emb,
                                   out_ema_cs, out_ema_w);
}
